// _ExactIntegerConv2d_26336739459125
// MI455X (gfx1250) — 8-bit matrix operand order repaired and checked by reading, compiled, not run on hardware
//
#include <hip/hip_runtime.h>
#include <stdint.h>

typedef int v8i __attribute__((ext_vector_type(8)));
typedef int v4i __attribute__((ext_vector_type(4)));

#define BQ    32
#define CINQ  256
#define COUTQ 256
#define HHQ   56
#define WWQ   56
#define HWSZ  (HHQ * WWQ)
#define POSN  (BQ * HWSZ)
#define KDIM  (CINQ * 9)
#define KT    (KDIM / 64)
#define HP    58
#define WP    58
#define IMGP  (HP * WP * CINQ)
#define LDSPITCH 80
#define LDSBUF   (128 * LDSPITCH)

#if defined(__has_builtin)
#if __has_builtin(__builtin_amdgcn_global_load_async_to_lds_b128) && \
    __has_builtin(__builtin_amdgcn_s_wait_asynccnt)
#define USE_ASYNC_LDS 1
#endif
#endif
#ifndef USE_ASYNC_LDS
#define USE_ASYNC_LDS 0
#endif

#if USE_ASYNC_LDS
static __device__ __forceinline__ void async_cp16(const uint8_t* gsrc, uint8_t* ldst) {
  __builtin_amdgcn_global_load_async_to_lds_b128((v4i*)gsrc, (v4i*)ldst, 0, 0);
}
#endif

__global__ __launch_bounds__(256) void quantize_pack_x(
    const float* __restrict__ x, const float* __restrict__ sA,
    const float* __restrict__ sQ, const float* __restrict__ sZP,
    const float* __restrict__ sGA, uint8_t* __restrict__ xqp) {
  __shared__ uint8_t tile[64][68];
  const float act_s = sA[0], act_q = sQ[0], azp = sZP[0], ga = sGA[0];
  const float zp = azp * act_s / ga;
  const float hi = zp + act_q - act_s;
  const float inv_s = 1.0f / act_s;

  const int pos0 = blockIdx.x * 64;
  const int c0 = blockIdx.y * 64;
  const int t = threadIdx.x;

#pragma unroll 4
  for (int cc = 0; cc < 16; ++cc) {
    int c = cc * 4 + (t >> 6);
    int p = pos0 + (t & 63);
    int b = p / HWSZ, hw = p % HWSZ;
    float v = x[(size_t)(b * CINQ + c0 + c) * HWSZ + hw];
    v = fminf(fmaxf(v, zp), hi);
    float code = ga * rintf((v - zp) * inv_s);
    tile[c][t & 63] = (uint8_t)(int)code;
  }
  __syncthreads();
#pragma unroll 4
  for (int pp = 0; pp < 16; ++pp) {
    int p = pos0 + pp * 4 + (t >> 6);
    int b = p / HWSZ, hw = p % HWSZ;
    int h = hw / WWQ, w = hw % WWQ;
    int c = t & 63;
    xqp[(size_t)b * IMGP + (size_t)((h + 1) * WP + (w + 1)) * CINQ + c0 + c] =
        tile[c][pp * 4 + (t >> 6)];
  }
}

__global__ __launch_bounds__(256) void zero_pad_border(uint8_t* __restrict__ xqp) {
  const int PER_IMG = 228 * 64;
  int tid = blockIdx.x * 256 + threadIdx.x;
  if (tid >= BQ * PER_IMG) return;
  int b = tid / PER_IMG;
  int r = tid % PER_IMG;
  int q = r >> 6, d = r & 63;
  int h, w;
  if (q < 58) { h = 0; w = q; }
  else if (q < 116) { h = HP - 1; w = q - 58; }
  else { int s = q - 116; h = 1 + (s >> 1); w = (s & 1) ? (WP - 1) : 0; }
  uint32_t* p = (uint32_t*)(xqp + (size_t)b * IMGP + (size_t)(h * WP + w) * CINQ);
  p[d] = 0u;
}

__global__ __launch_bounds__(256) void pack_weights(
    const float* __restrict__ wgt, const float* __restrict__ wzp,
    const float* __restrict__ sGW, int8_t* __restrict__ bfrag) {
  int idx = blockIdx.x * 256 + threadIdx.x;
  if (idx >= COUTQ * KDIM) return;
  int cout = idx / KDIM;
  int k = idx % KDIM;
  int patch = k >> 8;
  int cin = k & 255;
  int kh = patch / 3, kw = patch % 3;
  float v = sGW[0] * wgt[(((size_t)cout * CINQ + cin) * 3 + kh) * 3 + kw] + wzp[cout];
  int8_t q = (int8_t)(int)rintf(v);

  int nT = cout >> 4, col = cout & 15;
  int kt = k >> 6, kin = k & 63;
  int g = kin >> 5, rem = kin & 31;
  int khalf = rem >> 4, r2 = rem & 15;
  int lane = khalf * 16 + col;
  int vreg = g * 4 + (r2 >> 2);
  int byte = r2 & 3;
  bfrag[(((size_t)nT * KT + kt) * 32 + lane) * 32 + vreg * 4 + byte] = q;
}

static __device__ __forceinline__ v8i load_bfrag(const int8_t* p) {
  const v4i* q = (const v4i*)p;
  v4i lo = q[0], hi = q[1];
  v8i r;
  r[0] = lo[0]; r[1] = lo[1]; r[2] = lo[2]; r[3] = lo[3];
  r[4] = hi[0]; r[5] = hi[1]; r[6] = hi[2]; r[7] = hi[3];
  return r;
}

__global__ __launch_bounds__(256, 2) void conv_gemm_iu8(
    const uint8_t* __restrict__ xq, const int8_t* __restrict__ bfrag,
    const float* __restrict__ fp_scale, const float* __restrict__ bias,
    float* __restrict__ out) {
  __shared__ __align__(16) uint8_t smA[2 * LDSBUF];
  const int nBase = blockIdx.x * 64;
  const int mBase = blockIdx.y * 128;
  const int t = threadIdx.x;
  const int lane = t & 31, wv = t >> 5;

  v8i acc0 = (v8i)(0), acc1 = (v8i)(0), acc2 = (v8i)(0), acc3 = (v8i)(0);

  const int srow = t >> 1, sseg = t & 1;
  const int sm = mBase + srow;
  const int sb = sm / HWSZ, shw = sm % HWSZ;
  const int sh = shw / WWQ, sw = shw % WWQ;
  const uint8_t* pbase =
      xq + (size_t)sb * IMGP + (size_t)(sh * WP + sw) * CINQ + sseg * 32;
  uint8_t* ldst0 = smA + srow * LDSPITCH + sseg * 32;

  const int rowA = wv * 16 + (lane & 15);
  const int halfL = lane >> 4;
  const int8_t* bb = bfrag + ((size_t)(nBase >> 4) * KT) * 1024 + lane * 32;

#if USE_ASYNC_LDS
  async_cp16(pbase, ldst0);
  async_cp16(pbase + 16, ldst0 + 16);
  __builtin_amdgcn_s_wait_asynccnt(0);
#else
  {
    const uint4* s = (const uint4*)pbase;
    uint4 a0 = s[0], a1 = s[1];
    ((uint4*)ldst0)[0] = a0;
    ((uint4*)ldst0)[1] = a1;
  }
#endif
  __syncthreads();

#pragma unroll 1
  for (int kt = 0; kt < KT; ++kt) {
    const int cur = kt & 1;
    const int ktn = (kt + 1 < KT) ? (kt + 1) : kt;
    const int patchN = ktn >> 2;
    const int tapOffN =
        ((patchN / 3) * WP + (patchN % 3)) * CINQ + (ktn & 3) * 64;
    const uint8_t* nsrc = pbase + tapOffN;
    uint8_t* ndst = ldst0 + (cur ^ 1) * LDSBUF;

#if USE_ASYNC_LDS
    async_cp16(nsrc, ndst);
    async_cp16(nsrc + 16, ndst + 16);
#else
    const uint4* s = (const uint4*)nsrc;
    uint4 pf0 = s[0], pf1 = s[1];
#endif

    const int8_t* bk = bb + (size_t)kt * 1024;
    v8i b0 = load_bfrag(bk);
    v8i b1 = load_bfrag(bk + 1 * KT * 1024);
    v8i b2 = load_bfrag(bk + 2 * KT * 1024);
    v8i b3 = load_bfrag(bk + 3 * KT * 1024);

    const uint32_t* smc = (const uint32_t*)(smA + cur * LDSBUF);
    v8i afrag;
#pragma unroll
    for (int r = 0; r < 8; ++r) {
      int koff = ((r & 4) << 3) + (halfL << 4) + ((r & 3) << 2);
      afrag[r] = (int)smc[rowA * (LDSPITCH / 4) + (koff >> 2)];
    }

    acc0 = __builtin_amdgcn_wmma_i32_16x16x64_iu8(false, afrag, true, b0, acc0,
                                                  false, false);
    acc1 = __builtin_amdgcn_wmma_i32_16x16x64_iu8(false, afrag, true, b1, acc1,
                                                  false, false);
    acc2 = __builtin_amdgcn_wmma_i32_16x16x64_iu8(false, afrag, true, b2, acc2,
                                                  false, false);
    acc3 = __builtin_amdgcn_wmma_i32_16x16x64_iu8(false, afrag, true, b3, acc3,
                                                  false, false);

#if USE_ASYNC_LDS
    __builtin_amdgcn_s_wait_asynccnt(0);
#else
    ((uint4*)ndst)[0] = pf0;
    ((uint4*)ndst)[1] = pf1;
#endif
    __syncthreads();
  }

  const int col = lane & 15;
  const int mRow0 = mBase + wv * 16 + halfL * 8;
  v8i accs[4] = {acc0, acc1, acc2, acc3};
#pragma unroll
  for (int nSub = 0; nSub < 4; ++nSub) {
    int n = nBase + nSub * 16 + col;
    float sc = fp_scale[n], bi = bias[n];
#pragma unroll
    for (int r = 0; r < 8; ++r) {
      int m = mRow0 + r;
      int b = m / HWSZ, hw = m % HWSZ;
      out[(size_t)(b * COUTQ + n) * HWSZ + hw] = (float)accs[nSub][r] * sc + bi;
    }
  }
}

extern "C" void kernel_launch(void* const* d_in, const int* in_sizes, int n_in,
                              void* d_out, int out_size, void* d_ws, size_t ws_size,
                              hipStream_t stream) {
  (void)in_sizes; (void)n_in; (void)out_size; (void)ws_size;
  const float* x        = (const float*)d_in[0];
  const float* wgt      = (const float*)d_in[1];
  const float* act_s    = (const float*)d_in[2];
  const float* act_q    = (const float*)d_in[3];
  const float* azp      = (const float*)d_in[4];
  const float* wzp      = (const float*)d_in[5];
  const float* fp_scale = (const float*)d_in[6];
  const float* bias     = (const float*)d_in[7];
  const float* ga       = (const float*)d_in[8];
  const float* gw       = (const float*)d_in[9];
  float* out = (float*)d_out;

  uint8_t* xqp = (uint8_t*)d_ws;
  int8_t* bfrag = (int8_t*)d_ws + (size_t)BQ * IMGP;

  zero_pad_border<<<(BQ * 228 * 64 + 255) / 256, 256, 0, stream>>>(xqp);
  quantize_pack_x<<<dim3(POSN / 64, CINQ / 64), 256, 0, stream>>>(
      x, act_s, act_q, azp, ga, xqp);
  pack_weights<<<(COUTQ * KDIM + 255) / 256, 256, 0, stream>>>(wgt, wzp, gw, bfrag);
  conv_gemm_iu8<<<dim3(COUTQ / 64, POSN / 128), 256, 0, stream>>>(
      xqp, bfrag, fp_scale, bias, out);
}
